// AttentivePredictionFusion_50826642981243
// MI455X (gfx1250) — matrix operand order repaired and checked by reading, not compiled, not run on hardware
//
#include <hip/hip_runtime.h>
#include <hip/hip_bf16.h>

#define BB 8
#define TT 2048
#define DD 512
#define HH 128
#define MTOT (BB * TT) // 16384

typedef __attribute__((ext_vector_type(16))) _Float16 v16h;
typedef __attribute__((ext_vector_type(8)))  _Float16 v8h;
typedef __attribute__((ext_vector_type(8)))  float    v8f;
typedef __attribute__((ext_vector_type(4)))  float    v4f;

static __device__ __forceinline__ v8f wmma16(v16h a, v16h b, v8f c) {
  // v_wmma_f32_16x16x32_f16: D = A(16x32) * B(32x16) + C(16x16 f32)
  return __builtin_amdgcn_wmma_f32_16x16x32_f16(false, a, false, b, (short)0, c,
                                                false, false);
}

// A fragment 16x32 f16 (row-major source, ld in elements).
// Lane half h: K = {h*8..h*8+7} U {16+h*8..16+h*8+7}; lane&15 = row.
static __device__ __forceinline__ v16h frag_a_f16(const _Float16* base, int ld, int lane) {
  int h = lane >> 4, r = lane & 15;
  const _Float16* p = base + r * ld + h * 8;
  v8h lo = *(const v8h*)p;
  v8h hi = *(const v8h*)(p + 16);
  return __builtin_shufflevector(lo, hi, 0,1,2,3,4,5,6,7,8,9,10,11,12,13,14,15);
}

// B fragment 32x16 f16 from Bt stored [N][K] row-major.
// Lane half h: K = {h*8..h*8+7} U {16+h*8..16+h*8+7}; lane&15 = column n.
static __device__ __forceinline__ v16h frag_b_f16(const _Float16* base, int ld, int lane) {
  int h = lane >> 4, n = lane & 15;
  const _Float16* p = base + n * ld + h * 8;
  v8h lo = *(const v8h*)p;
  v8h hi = *(const v8h*)(p + 16);
  return __builtin_shufflevector(lo, hi, 0,1,2,3,4,5,6,7,8,9,10,11,12,13,14,15);
}

// ---- prep kernels -----------------------------------------------------------

// W[K][N] fp32 -> Wt[N][K] f16
__global__ void apf_transpose_cvt(const float* __restrict__ W, _Float16* __restrict__ Wt,
                                  int K, int N) {
  long idx = (long)blockIdx.x * blockDim.x + threadIdx.x;
  if (idx >= (long)K * N) return;
  int k = (int)(idx / N), n = (int)(idx % N);
  Wt[(long)n * K + k] = (_Float16)W[idx];
}

// prediction fp32 [MTOT][DD] -> first half of fusion buffer f16 [MTOT][2*DD]
__global__ void apf_cvt_pred(const float* __restrict__ pred, _Float16* __restrict__ F) {
  long idx = (long)blockIdx.x * blockDim.x + threadIdx.x;
  long row = idx / DD, c = idx % DD;
  F[row * (2 * DD) + c] = (_Float16)pred[idx];
}

// dense fp32 -> f16 copy (for x)
__global__ void apf_cvt_f16(const float* __restrict__ src, _Float16* __restrict__ dst) {
  long idx = (long)blockIdx.x * blockDim.x + threadIdx.x;
  dst[idx] = (_Float16)src[idx];
}

// ---- all-f16 GEMM: C[M,N] = A_f16[M,K](ld=lda) @ Bt_f16[N,K]^T + bias ------
// Block tile 128x128, wave tile 32x64 (8 wmma / K-step).
// transC=0: C f16 row-major [M][N]; transC=1: C f16 [n][ldCt] (for vT)
__global__ __launch_bounds__(256) void apf_gemm_f16(
    const _Float16* __restrict__ A, int lda,
    const _Float16* __restrict__ Bt, const float* __restrict__ bias,
    _Float16* __restrict__ C, int K, int N, int transC, int ldCt) {
  int lane = threadIdx.x & 31;
  int w = threadIdx.x >> 5;
  int wm = w >> 1, wn = w & 1;           // 4x2 wave grid
  long m0 = (long)blockIdx.x * 128 + wm * 32;
  int n0 = blockIdx.y * 128 + wn * 64;
  v8f acc[2][4] = {};
  for (int k = 0; k < K; k += 32) {
    v16h a0 = frag_a_f16(A + m0 * lda + k, lda, lane);
    v16h a1 = frag_a_f16(A + (m0 + 16) * lda + k, lda, lane);
    v16h bfr[4];
#pragma unroll
    for (int j = 0; j < 4; ++j)
      bfr[j] = frag_b_f16(Bt + (long)(n0 + j * 16) * K + k, K, lane);
#pragma unroll
    for (int j = 0; j < 4; ++j) {
      acc[0][j] = wmma16(a0, bfr[j], acc[0][j]);
      acc[1][j] = wmma16(a1, bfr[j], acc[1][j]);
    }
  }
  int h = lane >> 4, r = lane & 15;
#pragma unroll
  for (int i2 = 0; i2 < 2; ++i2)
#pragma unroll
    for (int j2 = 0; j2 < 4; ++j2) {
      long gm = m0 + i2 * 16 + 8 * h;
      int gn = n0 + j2 * 16 + r;
      float bb = bias[gn];
      if (!transC) {
#pragma unroll
        for (int i = 0; i < 8; ++i)
          C[(gm + i) * N + gn] = (_Float16)(acc[i2][j2][i] + bb);
      } else {
        v8h pk;
#pragma unroll
        for (int i = 0; i < 8; ++i) pk[i] = (_Float16)(acc[i2][j2][i] + bb);
        *(v8h*)(C + (long)gn * ldCt + gm) = pk;
      }
    }
}

// ---- flash attention: softmax(q k^T) v, writes f16 into fusion buffer ------
// q,k: f16 [B*T][H] row-major. vT: f16 [D][B*T] (transposed). F: [B*T][2D].
__global__ __launch_bounds__(256) void apf_flash_attn(
    const _Float16* __restrict__ q, const _Float16* __restrict__ k,
    const _Float16* __restrict__ vT, _Float16* __restrict__ F) {
  __shared__ float    S[64 * 68];       // score chunk fp32
  __shared__ _Float16 P[64 * 88];       // exp probs f16 (ld=88 -> 16B-aligned rows)
  __shared__ float    mrow[64], lrow[64], crow[64];

  int tid = threadIdx.x;
  int lane = tid & 31, w = tid >> 5;
  int b = blockIdx.y;
  int qr0 = blockIdx.x * 64;
  const _Float16* qbase = q + (long)(b * TT + qr0) * HH;
  const _Float16* kbase = k + (long)b * TT * HH;
  const _Float16* vbase = vT + (long)b * TT;   // column offset into [D][MTOT]
  int d0 = w * 64;                              // this wave's D-slice

  if (tid < 64) { mrow[tid] = -1e30f; lrow[tid] = 0.f; }

  v8f O[4][4] = {};                             // 64 rows x 64 d per wave
  int smt = w >> 1;                             // score M-tile for phase 1
  int snt0 = (w & 1) * 2;                       // score N-tile pair

  for (int sc = 0; sc < TT; sc += 64) {
    // prefetch next chunk of K and V toward this WGP (global_prefetch_b8)
    if (sc + 64 < TT) {
      __builtin_prefetch(kbase + (long)(sc + 64 + (tid & 63)) * HH + (tid >> 6) * 32, 0, 1);
      __builtin_prefetch(vbase + (long)(tid * 2) * MTOT + sc + 64, 0, 1);
      __builtin_prefetch(vbase + (long)(tid * 2 + 1) * MTOT + sc + 64, 0, 1);
    }
    // ---- phase 1: S = Q(64x128) * Kchunk^T(128x64) ----
    v8f s0 = {}, s1 = {};
#pragma unroll
    for (int kk = 0; kk < HH; kk += 32) {
      v16h af = frag_a_f16(qbase + smt * 16 * HH + kk, HH, lane);
      v16h b0 = frag_b_f16(kbase + (long)(sc + snt0 * 16) * HH + kk, HH, lane);
      v16h b1 = frag_b_f16(kbase + (long)(sc + (snt0 + 1) * 16) * HH + kk, HH, lane);
      s0 = wmma16(af, b0, s0);
      s1 = wmma16(af, b1, s1);
    }
    {
      int h = lane >> 4, r = lane & 15;
      int row = smt * 16 + 8 * h;
#pragma unroll
      for (int i = 0; i < 8; ++i) {
        S[(row + i) * 68 + snt0 * 16 + r]       = s0[i];
        S[(row + i) * 68 + (snt0 + 1) * 16 + r] = s1[i];
      }
    }
    __syncthreads();
    // ---- phase 2: online softmax update (4 threads per row) ----
    {
      int row = tid >> 2, seg = tid & 3;
      const float* Sr = &S[row * 68 + seg * 16];
      float mloc = -1e30f;
#pragma unroll
      for (int j = 0; j < 16; ++j) mloc = fmaxf(mloc, Sr[j]);
      mloc = fmaxf(mloc, __shfl_xor(mloc, 1));
      mloc = fmaxf(mloc, __shfl_xor(mloc, 2));
      float mold = mrow[row];
      float mnew = fmaxf(mold, mloc);
      float sum = 0.f;
      _Float16* Pr = &P[row * 88 + seg * 16];
#pragma unroll
      for (int j = 0; j < 16; ++j) {
        float pv = __expf(Sr[j] - mnew);
        Pr[j] = (_Float16)pv;
        sum += pv;
      }
      sum += __shfl_xor(sum, 1);
      sum += __shfl_xor(sum, 2);
      if (seg == 0) {
        float c = __expf(mold - mnew);
        crow[row] = c;
        lrow[row] = lrow[row] * c + sum;
        mrow[row] = mnew;
      }
    }
    __syncthreads();
    // ---- phase 3: rescale O, then O += P(64x64) * Vchunk(64 x Dslice64) ----
    {
      int h = lane >> 4;
#pragma unroll
      for (int mt = 0; mt < 4; ++mt) {
        float cc[8];
#pragma unroll
        for (int i = 0; i < 8; ++i) cc[i] = crow[mt * 16 + 8 * h + i];
#pragma unroll
        for (int nt = 0; nt < 4; ++nt)
#pragma unroll
          for (int i = 0; i < 8; ++i) O[mt][nt][i] *= cc[i];
      }
    }
#pragma unroll
    for (int kk = 0; kk < 64; kk += 32) {
      v16h bf[4];
#pragma unroll
      for (int nt = 0; nt < 4; ++nt)
        bf[nt] = frag_b_f16(vbase + (long)(d0 + nt * 16) * MTOT + sc + kk, MTOT, lane);
#pragma unroll
      for (int mt = 0; mt < 4; ++mt) {
        v16h af = frag_a_f16(&P[mt * 16 * 88 + kk], 88, lane);
#pragma unroll
        for (int nt = 0; nt < 4; ++nt) O[mt][nt] = wmma16(af, bf[nt], O[mt][nt]);
      }
    }
  }
  // ---- epilogue: O / l  -> second half of fusion buffer ----
  {
    int h = lane >> 4, r = lane & 15;
#pragma unroll
    for (int mt = 0; mt < 4; ++mt) {
      float inv[8];
#pragma unroll
      for (int i = 0; i < 8; ++i) inv[i] = 1.f / lrow[mt * 16 + 8 * h + i];
#pragma unroll
      for (int nt = 0; nt < 4; ++nt)
#pragma unroll
        for (int i = 0; i < 8; ++i) {
          long row = (long)b * TT + qr0 + mt * 16 + 8 * h + i;
          int col = DD + d0 + nt * 16 + r;
          F[row * (2 * DD) + col] = (_Float16)(O[mt][nt][i] * inv[i]);
        }
    }
  }
}

// ---- fusion GEMM: out = sigmoid(F[M,1024] @ Wf + bf), fp32 out -------------
// Block tile 128x128, wave tile 32x64.
__global__ __launch_bounds__(256) void apf_gemm_fusion(
    const _Float16* __restrict__ Fb, const _Float16* __restrict__ Wft,
    const float* __restrict__ bf, float* __restrict__ out) {
  const int K = 2 * DD;
  int lane = threadIdx.x & 31;
  int w = threadIdx.x >> 5;
  int wm = w >> 1, wn = w & 1;
  long m0 = (long)blockIdx.x * 128 + wm * 32;
  int n0 = blockIdx.y * 128 + wn * 64;
  v8f acc[2][4] = {};
  for (int k = 0; k < K; k += 32) {
    v16h a0 = frag_a_f16(Fb + m0 * K + k, K, lane);
    v16h a1 = frag_a_f16(Fb + (m0 + 16) * K + k, K, lane);
    v16h bfr[4];
#pragma unroll
    for (int j = 0; j < 4; ++j)
      bfr[j] = frag_b_f16(Wft + (long)(n0 + j * 16) * K + k, K, lane);
#pragma unroll
    for (int j = 0; j < 4; ++j) {
      acc[0][j] = wmma16(a0, bfr[j], acc[0][j]);
      acc[1][j] = wmma16(a1, bfr[j], acc[1][j]);
    }
  }
  int h = lane >> 4, r = lane & 15;
#pragma unroll
  for (int i2 = 0; i2 < 2; ++i2)
#pragma unroll
    for (int j2 = 0; j2 < 4; ++j2) {
      long gm = m0 + i2 * 16 + 8 * h;
      int gn = n0 + j2 * 16 + r;
      float bb = bf[gn];
#pragma unroll
      for (int i = 0; i < 8; ++i) {
        float vacc = acc[i2][j2][i] + bb;
        out[(gm + i) * DD + gn] = 1.f / (1.f + __expf(-vacc));
      }
    }
}

extern "C" void kernel_launch(void* const* d_in, const int* in_sizes, int n_in,
                              void* d_out, int out_size, void* d_ws, size_t ws_size,
                              hipStream_t stream) {
  (void)in_sizes; (void)n_in; (void)out_size; (void)ws_size;
  const float* x    = (const float*)d_in[0];
  const float* pred = (const float*)d_in[1];
  const float* Wq   = (const float*)d_in[2];
  const float* bq   = (const float*)d_in[3];
  const float* Wk   = (const float*)d_in[4];
  const float* bk   = (const float*)d_in[5];
  const float* Wv   = (const float*)d_in[6];
  const float* bv   = (const float*)d_in[7];
  const float* Wf   = (const float*)d_in[8];
  const float* bf   = (const float*)d_in[9];
  float* out = (float*)d_out;

  char* ws = (char*)d_ws;
  _Float16* qf  = (_Float16*)ws; ws += (size_t)MTOT * HH * 2;      // 4 MB
  _Float16* kf  = (_Float16*)ws; ws += (size_t)MTOT * HH * 2;      // 4 MB
  _Float16* vT  = (_Float16*)ws; ws += (size_t)DD * MTOT * 2;      // 16 MB
  _Float16* F   = (_Float16*)ws; ws += (size_t)MTOT * 2 * DD * 2;  // 32 MB
  _Float16* xh  = (_Float16*)ws; ws += (size_t)MTOT * DD * 2;      // 16 MB
  _Float16* Wqt = (_Float16*)ws; ws += (size_t)HH * DD * 2;
  _Float16* Wkt = (_Float16*)ws; ws += (size_t)HH * DD * 2;
  _Float16* Wvt = (_Float16*)ws; ws += (size_t)DD * DD * 2;
  _Float16* Wft = (_Float16*)ws; ws += (size_t)DD * 2 * DD * 2;

  apf_transpose_cvt<<<(DD * HH + 255) / 256, 256, 0, stream>>>(Wq, Wqt, DD, HH);
  apf_transpose_cvt<<<(DD * HH + 255) / 256, 256, 0, stream>>>(Wk, Wkt, DD, HH);
  apf_transpose_cvt<<<(DD * DD + 255) / 256, 256, 0, stream>>>(Wv, Wvt, DD, DD);
  apf_transpose_cvt<<<(2 * DD * DD + 255) / 256, 256, 0, stream>>>(Wf, Wft, 2 * DD, DD);
  apf_cvt_pred<<<(MTOT * DD) / 256, 256, 0, stream>>>(pred, F);
  apf_cvt_f16<<<(MTOT * DD) / 256, 256, 0, stream>>>(x, xh);

  // q = pred @ Wq + bq  (A = first half of fusion buffer, already f16, ld=1024)
  apf_gemm_f16<<<dim3(MTOT / 128, HH / 128), 256, 0, stream>>>(
      F, 2 * DD, Wqt, bq, qf, DD, HH, 0, 0);
  // k = x @ Wk + bk
  apf_gemm_f16<<<dim3(MTOT / 128, HH / 128), 256, 0, stream>>>(
      xh, DD, Wkt, bk, kf, DD, HH, 0, 0);
  // v = x @ Wv + bv, stored transposed as vT[D][B*T]
  apf_gemm_f16<<<dim3(MTOT / 128, DD / 128), 256, 0, stream>>>(
      xh, DD, Wvt, bv, vT, DD, DD, 1, MTOT);

  apf_flash_attn<<<dim3(TT / 64, BB), 256, 0, stream>>>(qf, kf, vT, F);

  apf_gemm_fusion<<<dim3(MTOT / 128, DD / 128), 256, 0, stream>>>(F, Wft, bf, out);
}
